// GraphNetworkLayer_16045997817971
// MI455X (gfx1250) — matrix operand order repaired and checked by reading, not compiled, not run on hardware
//
#include <hip/hip_runtime.h>

// GNN layer for MI455X (gfx1250), wave32 WMMA f16 (f32 accumulate):
//   new_edges = relu([edges | nodes[snd] | nodes[rcv]] @ W_edge + b)  78.6 GFLOP
//   received  = segment_sum(new_edges, receivers)   (coalesced f32 atomics, L2-resident)
//   new_nodes = relu([nodes | received] @ W_node + b)
// LDS tiles stored in exact WMMA per-lane fragment order -> conflict-free ds_load_b128,
// W staged to LDS with global_load_async_to_lds_b128 (ASYNCcnt), streaming data uses NT hints.

#define DFEAT 64
#define DOUT  128
#define DIN   192
#define NNODES 50000
#define NEDGES 1600000
#define THREADS 256
#define NWAVES 8
#define KSTEPS 6                   // 192 / 32
#define FRAG   16                  // f16 per lane per fragment
#define MT_EDGE 2                  // M-tiles (of 16 rows) per wave in edge kernel
#define EDGE_ROWS_PER_BLOCK (NWAVES * MT_EDGE * 16)          // 256
#define EDGE_BLOCKS (NEDGES / EDGE_ROWS_PER_BLOCK)           // 6250
#define NODE_TILES (NNODES / 16)                             // 3125
#define ATILE_F16 (KSTEPS * 32 * FRAG)                       // 3072 f16 per 16-row M-tile
#define WMAT_F16  (8 * KSTEPS * 32 * FRAG)                   // 24576 f16 (48 KB)

typedef __attribute__((ext_vector_type(16))) _Float16 v16h;
typedef __attribute__((ext_vector_type(8)))  _Float16 v8h;
typedef __attribute__((ext_vector_type(8)))  float    v8f;
typedef __attribute__((ext_vector_type(4)))  float    f32x4;

union V16 { v16h v; v8h h[2]; };
union H4  { _Float16 h[4]; uint2 u; };

// Map a 4-wide K chunk (row r, K0=4*c4) into the A-fragment f16 offset within one M-tile.
// Frag layout: chunk (ks, lane) holds 16 f16: lower lanes K {ks*32+0..7, +16..23},
// upper lanes (lane+16) K {ks*32+8..15, +24..31}; N/A-row = lane&15.
__device__ __forceinline__ int frag_off(int r, int K0) {
  int ks  = K0 >> 5;
  int rem = K0 & 31;
  int hi2 = rem >> 3;                       // 0..3
  int lp  = (r & 15) + ((hi2 & 1) << 4);    // fragment lane
  return (ks * 32 + lp) * FRAG + ((hi2 >> 1) << 3) + (rem & 7);
}

// ---- prep: W f32[K][N] -> f16 in B-fragment order: [t][ks][lane][16] ----
__global__ void prep_weights(const float* __restrict__ We,
                             const float* __restrict__ Wn,
                             _Float16* __restrict__ WtE,
                             _Float16* __restrict__ WtN) {
  int j = blockIdx.x * blockDim.x + threadIdx.x;
  if (j >= 2 * WMAT_F16) return;
  int m = j / WMAT_F16;
  int o = j % WMAT_F16;
  int i = o & 15;
  int l = (o >> 4) & 31;
  int g = o >> 9;                 // (t*6 + ks)
  int ks = g % KSTEPS, t = g / KSTEPS;
  int N = t * 16 + (l & 15);
  int K = ks * 32 + ((i >> 3) << 4) + ((l >> 4) << 3) + (i & 7);    // upper lanes hold K+8..15, +24..31
  const float* W = m ? Wn : We;
  _Float16*   Wt = m ? WtN : WtE;
  Wt[o] = (_Float16)W[K * DOUT + N];
}

__global__ void zero_f32(float* __restrict__ p, int n4) {
  int i = blockIdx.x * blockDim.x + threadIdx.x;
  if (i < n4) ((f32x4*)p)[i] = (f32x4)(0.f);
}

// Async 48KB linear stage of fragment-ordered W into LDS (ASYNCcnt-tracked).
__device__ __forceinline__ void stage_W(const _Float16* __restrict__ Wt,
                                        _Float16* __restrict__ sW, int tid) {
  const uint4* gsrc = (const uint4*)Wt;
  unsigned lbase = (unsigned)(uintptr_t)(void*)sW;
  #pragma unroll
  for (int i = 0; i < (WMAT_F16 * 2 / 16) / THREADS; ++i) {   // 12 x b128 per thread
    unsigned dst = lbase + (unsigned)(i * THREADS + tid) * 16u;
    const uint4* src = gsrc + (i * THREADS + tid);
    asm volatile("global_load_async_to_lds_b128 %0, %1, off"
                 :: "v"(dst), "v"(src) : "memory");
  }
  asm volatile("s_wait_asynccnt 0" ::: "memory");
}

// ---------------- edge MLP + fused atomic segment-sum ----------------
__global__ __launch_bounds__(THREADS)
void edge_kernel(const float* __restrict__ nodes,
                 const float* __restrict__ edges,
                 const int*   __restrict__ senders,
                 const int*   __restrict__ receivers,
                 const _Float16* __restrict__ Wt,
                 const float* __restrict__ bias,
                 float* __restrict__ out_edges,
                 float* __restrict__ received) {
  __shared__ __align__(16) _Float16 sW[WMAT_F16];                      // 48 KB
  __shared__ __align__(16) _Float16 sA[NWAVES][MT_EDGE * ATILE_F16];   // 96 KB

  const int tid  = threadIdx.x;
  stage_W(Wt, sW, tid);

  const int wave = tid >> 5, lane = tid & 31;
  const int base = (blockIdx.x * NWAVES + wave) * (MT_EDGE * 16);      // first edge row

  // Gather: lane owns one of the wave's 32 edge rows (mt = lane>>4).
  _Float16* A = sA[wave];
  {
    const int e    = base + lane;
    const int sidx = senders[e];
    const int ridx = receivers[e];
    const int moff = (lane >> 4) * ATILE_F16;
    const int r    = lane & 15;
    const f32x4* pe = (const f32x4*)(edges + (size_t)e    * DFEAT);
    const f32x4* ps = (const f32x4*)(nodes + (size_t)sidx * DFEAT);
    const f32x4* pr = (const f32x4*)(nodes + (size_t)ridx * DFEAT);
    #pragma unroll
    for (int c4 = 0; c4 < 48; ++c4) {
      f32x4 f;
      if (c4 < 16)      f = __builtin_nontemporal_load(pe + c4);  // streaming, keep L2 free
      else if (c4 < 32) f = ps[c4 - 16];
      else              f = pr[c4 - 32];
      H4 pk;
      pk.h[0] = (_Float16)f.x; pk.h[1] = (_Float16)f.y;
      pk.h[2] = (_Float16)f.z; pk.h[3] = (_Float16)f.w;
      *(uint2*)&A[moff + frag_off(r, c4 * 4)] = pk.u;
    }
  }
  __syncthreads();

  // Compute: 2 M-tiles share every B fragment. All LDS reads conflict-free b128.
  v8f acc[MT_EDGE][8] = {};
  const _Float16* Af = A  + lane * FRAG;
  const _Float16* Bf = sW + lane * FRAG;
  #pragma unroll
  for (int ks = 0; ks < KSTEPS; ++ks) {
    V16 a0, a1;
    a0.h[0] = *(const v8h*)(Af + ks * 512);
    a0.h[1] = *(const v8h*)(Af + ks * 512 + 8);
    a1.h[0] = *(const v8h*)(Af + ATILE_F16 + ks * 512);
    a1.h[1] = *(const v8h*)(Af + ATILE_F16 + ks * 512 + 8);
    #pragma unroll
    for (int t = 0; t < 8; ++t) {
      V16 b;
      b.h[0] = *(const v8h*)(Bf + (t * KSTEPS + ks) * 512);
      b.h[1] = *(const v8h*)(Bf + (t * KSTEPS + ks) * 512 + 8);
      acc[0][t] = __builtin_amdgcn_wmma_f32_16x16x32_f16(
          false, a0.v, false, b.v, (short)0, acc[0][t], false, false);
      acc[1][t] = __builtin_amdgcn_wmma_f32_16x16x32_f16(
          false, a1.v, false, b.v, (short)0, acc[1][t], false, false);
    }
  }

  // Epilogue: bias+relu, NT store of new_edges, fused coalesced f32 atomics.
  const int l15 = lane & 15, Mbase = (lane >> 4) * 8;
  #pragma unroll
  for (int mt = 0; mt < MT_EDGE; ++mt) {
    const int rowbase = base + mt * 16 + Mbase;
    int rcv[8];
    #pragma unroll
    for (int i = 0; i < 8; ++i) rcv[i] = receivers[rowbase + i];
    #pragma unroll
    for (int t = 0; t < 8; ++t) {
      int N = t * 16 + l15;
      float bv = bias[N];
      #pragma unroll
      for (int i = 0; i < 8; ++i) {
        float v = acc[mt][t][i] + bv;
        v = v > 0.f ? v : 0.f;
        __builtin_nontemporal_store(v, &out_edges[(size_t)(rowbase + i) * DOUT + N]);
        __hip_atomic_fetch_add(&received[(size_t)rcv[i] * DOUT + N], v,
                               __ATOMIC_RELAXED, __HIP_MEMORY_SCOPE_AGENT);
      }
    }
  }
}

// ---------------- node MLP ----------------
__global__ __launch_bounds__(THREADS)
void node_kernel(const float* __restrict__ nodes,
                 const float* __restrict__ received,
                 const _Float16* __restrict__ Wt,
                 const float* __restrict__ bias,
                 float* __restrict__ out_nodes) {
  __shared__ __align__(16) _Float16 sW[WMAT_F16];          // 48 KB
  __shared__ __align__(16) _Float16 sA[NWAVES][ATILE_F16]; // 48 KB

  const int tid = threadIdx.x;
  stage_W(Wt, sW, tid);

  const int wave = tid >> 5, lane = tid & 31;
  const int tile = blockIdx.x * NWAVES + wave;
  const bool active = tile < NODE_TILES;

  _Float16* A = sA[wave];
  if (active) {
    const int r    = lane & 15;
    const int half = lane >> 4;               // two lanes share a row, split chunks
    const int node = tile * 16 + r;
    const f32x4* pn = (const f32x4*)(nodes    + (size_t)node * DFEAT);
    const f32x4* pa = (const f32x4*)(received + (size_t)node * DOUT);
    #pragma unroll
    for (int it = 0; it < 24; ++it) {
      int c4 = half * 24 + it;
      f32x4 f = (c4 < 16) ? pn[c4] : pa[c4 - 16];
      H4 pk;
      pk.h[0] = (_Float16)f.x; pk.h[1] = (_Float16)f.y;
      pk.h[2] = (_Float16)f.z; pk.h[3] = (_Float16)f.w;
      *(uint2*)&A[frag_off(r, c4 * 4)] = pk.u;
    }
  }
  __syncthreads();
  if (!active) return;

  v8f acc[8] = {};
  const _Float16* Af = A  + lane * FRAG;
  const _Float16* Bf = sW + lane * FRAG;
  #pragma unroll
  for (int ks = 0; ks < KSTEPS; ++ks) {
    V16 a;
    a.h[0] = *(const v8h*)(Af + ks * 512);
    a.h[1] = *(const v8h*)(Af + ks * 512 + 8);
    #pragma unroll
    for (int t = 0; t < 8; ++t) {
      V16 b;
      b.h[0] = *(const v8h*)(Bf + (t * KSTEPS + ks) * 512);
      b.h[1] = *(const v8h*)(Bf + (t * KSTEPS + ks) * 512 + 8);
      acc[t] = __builtin_amdgcn_wmma_f32_16x16x32_f16(
          false, a.v, false, b.v, (short)0, acc[t], false, false);
    }
  }

  const int l15 = lane & 15, Mbase = (lane >> 4) * 8;
  #pragma unroll
  for (int t = 0; t < 8; ++t) {
    int N = t * 16 + l15;
    float bv = bias[N];
    #pragma unroll
    for (int i = 0; i < 8; ++i) {
      float v = acc[t][i] + bv;
      v = v > 0.f ? v : 0.f;
      out_nodes[(size_t)(tile * 16 + Mbase + i) * DOUT + N] = v;
    }
  }
}

// ---------------- host ----------------
extern "C" void kernel_launch(void* const* d_in, const int* in_sizes, int n_in,
                              void* d_out, int out_size, void* d_ws, size_t ws_size,
                              hipStream_t stream) {
  const float* nodes     = (const float*)d_in[0];
  const float* edges     = (const float*)d_in[1];
  const int*   senders   = (const int*)d_in[2];
  const int*   receivers = (const int*)d_in[3];
  const float* W_edge    = (const float*)d_in[4];
  const float* b_edge    = (const float*)d_in[5];
  const float* W_node    = (const float*)d_in[6];
  const float* b_node    = (const float*)d_in[7];

  float* new_nodes = (float*)d_out;                         // [N,128] first
  float* new_edges = (float*)d_out + (size_t)NNODES * DOUT; // then [E,128]

  char* ws = (char*)d_ws;
  _Float16* WtE      = (_Float16*)ws;                       // 48 KB
  _Float16* WtN      = (_Float16*)(ws + WMAT_F16 * 2);      // 48 KB
  float*    received = (float*)   (ws + 2 * WMAT_F16 * 2);  // 25.6 MB

  prep_weights<<<(2 * WMAT_F16 + 255) / 256, 256, 0, stream>>>(W_edge, W_node, WtE, WtN);

  int n4 = NNODES * DOUT / 4;
  zero_f32<<<(n4 + 255) / 256, 256, 0, stream>>>(received, n4);

  edge_kernel<<<EDGE_BLOCKS, THREADS, 0, stream>>>(
      nodes, edges, senders, receivers, WtE, b_edge, new_edges, received);

  node_kernel<<<(NODE_TILES + NWAVES - 1) / NWAVES, THREADS, 0, stream>>>(
      nodes, received, WtN, b_node, new_nodes);
}
